// SupConNoMasking_63007170232511
// MI455X (gfx1250) — matrix operand order repaired and checked by reading, not compiled, not run on hardware
//
#include <hip/hip_runtime.h>
#include <math.h>

// ---------------------------------------------------------------------------
// SupCon (no masking) fused loss for MI455X / gfx1250.
//   wave32 + v_wmma_f32_16x16x32_bf16 + TDM (tensor_load_to_lds) staging.
//   embeddings [16,256,512] fp32 -> loss scalar fp32
//   feats = normalize(transpose) -> [8192,256] bf16 in d_ws
//   loss_i = S - sim[i,pos] + log( sum_j w_ij * exp(sim_ij - S) ),  S = 1/T
//   pos(i) = (i + 4096) mod 8192 = i ^ 4096
// ---------------------------------------------------------------------------

typedef __attribute__((ext_vector_type(16))) __bf16        v16bf;
typedef __attribute__((ext_vector_type(8)))  float         v8f;
typedef __attribute__((ext_vector_type(4)))  unsigned int  u32x4;
typedef __attribute__((ext_vector_type(4)))  unsigned int  v4u;
typedef __attribute__((ext_vector_type(4)))  int           v4i;
typedef __attribute__((ext_vector_type(8)))  int           v8i;

union Frag { v16bf v; u32x4 u[2]; };

#if defined(__has_builtin)
#  if __has_builtin(__builtin_amdgcn_tensor_load_to_lds)
#    define HAVE_TDM 1
#  endif
#endif
#ifndef HAVE_TDM
#  define HAVE_TDM 0
#endif

#define LDS_ROW    264   // 256 bf16 + 8 pad (TDM: 4 dwords per 128 dwords)
#define STAGE_ROWS 64    // B rows staged per TDM transfer (32 KB)
#define CSPLIT     4     // column-range split (4 strips of 2048 columns)
#define STAGES_PER_BLOCK 32   // 2048 / 64

// ---------------------------------------------------------------------------
// Kernel 1: transpose [C=256, L=512] slabs -> rows, L2-normalize, pack bf16.
// Grid: 256 blocks = 16 batch * 16 l-chunks of 32; block = 256 threads.
// ---------------------------------------------------------------------------
__global__ __launch_bounds__(256) void supcon_normalize(
    const float* __restrict__ emb,            // [16,256,512]
    unsigned short* __restrict__ feats)       // [8192,256] bf16 bits
{
  __shared__ float tile[256][33];
  __shared__ float part[8][32];
  __shared__ float scale[32];

  const int t  = threadIdx.x;
  const int b  = blockIdx.x >> 4;             // 0..15
  const int l0 = (blockIdx.x & 15) << 5;      // 0,32,..,480

  #pragma unroll 4
  for (int rep = 0; rep < 32; ++rep) {
    int idx = rep * 256 + t;
    int c   = idx >> 5;
    int lx  = idx & 31;
    tile[c][lx] = emb[((size_t)(b * 256 + c)) * 512 + (size_t)(l0 + lx)];
  }
  __syncthreads();

  {
    int l = t & 31, seg = t >> 5;
    float s = 0.f;
    #pragma unroll
    for (int k = 0; k < 32; ++k) {
      float x = tile[seg * 32 + k][l];
      s = fmaf(x, x, s);
    }
    part[seg][l] = s;
  }
  __syncthreads();
  if (t < 32) {
    float s = 0.f;
    #pragma unroll
    for (int seg = 0; seg < 8; ++seg) s += part[seg][t];
    scale[t] = 1.0f / fmaxf(sqrtf(s), 1e-12f);   // torch F.normalize eps
  }
  __syncthreads();

  const size_t rowbase = (b < 8) ? ((size_t)b * 512 + l0)
                                 : ((size_t)4096 + (size_t)(b - 8) * 512 + l0);
  #pragma unroll 4
  for (int l = 0; l < 32; ++l) {
    float v = tile[t][l] * scale[l];
    unsigned int bits = __float_as_uint(v);                    // RNE fp32->bf16
    unsigned int r = (bits + 0x7FFFu + ((bits >> 16) & 1u)) >> 16;
    feats[(rowbase + l) * 256 + t] = (unsigned short)r;
  }
}

// ---------------------------------------------------------------------------
// TDM: async-load a 64x256 bf16 tile (32KB) into LDS, 16B pad per 512B row.
// D# per CDNA5 ISA §8 (groups 0/1; groups 2/3 zero => 2D tensor).
// ---------------------------------------------------------------------------
__device__ __forceinline__ void tdm_load_stage(unsigned lds_off,
                                               unsigned long long gaddr) {
#if HAVE_TDM
  v4u g0 = { 1u,                                   // count=1 (valid D#)
             lds_off,                              // lds_addr (bytes)
             (unsigned)gaddr,                      // global_addr[31:0]
             (((unsigned)(gaddr >> 32)) & 0x01FFFFFFu) | (2u << 30) }; // |type=2
  // data_size=2B, pad_enable, pad_interval=6 (128 dwords), pad_amount=3 (4 dw)
  v8i g1 = { (int)((1u << 16) | (1u << 20) | (6u << 22) | (3u << 25)),
             (int)(256u << 16),                    // tensor_dim0 = 256
             (int)((unsigned)STAGE_ROWS << 16),    // tensor_dim1 = 64
             (int)(256u << 16),                    // tile_dim0   = 256
             STAGE_ROWS,                           // tile_dim1   = 64
             256,                                  // tensor_dim0_stride = 256
             0, 0 };
  v4i z4 = { 0, 0, 0, 0 };
#if __clang_major__ >= 23
  v8i z8 = { 0, 0, 0, 0, 0, 0, 0, 0 };
  __builtin_amdgcn_tensor_load_to_lds(g0, g1, z4, z4, z8, 0);
#else
  __builtin_amdgcn_tensor_load_to_lds(g0, g1, z4, z4, 0);
#endif
#endif
}

// ---------------------------------------------------------------------------
// Kernel 2: fused Gram + weighted softmax partial denominators.
// One wave owns TWO 16-row A stripes (M=32, A resident in 128 VGPRs): each
// B fragment feeds two back-to-back WMMAs (2x matrix work per ds_load pair).
// Blocks split the 8192-column range into 4 strips; the 8 waves of a block
// share a 64-column TDM-staged, LDS double-buffered B stream.
// Grid: 128 blocks (= 32 row-groups x 4 column strips) * 256 threads.
// Outputs (deterministic, no atomics):
//   parts[cs][row]  partial denominator for this strip
//   rowpos[row]     positive-pair cosine (written by exactly one strip)
// ---------------------------------------------------------------------------
__global__ __launch_bounds__(256) void supcon_wmma(
    const unsigned short* __restrict__ feats,  // [8192,256] bf16 bits
    float* __restrict__ rowpos,                // [8192]
    float* __restrict__ parts)                 // [CSPLIT][8192]
{
  __shared__ __align__(16) unsigned short bbuf[2][STAGE_ROWS * LDS_ROW];

  const int lane = threadIdx.x & 31;
  const int wib  = threadIdx.x >> 5;
  const int rg   = blockIdx.x >> 2;            // row group 0..31
  const int cs   = blockIdx.x & 3;             // column strip 0..3
  const int row_base  = (rg * 8 + wib) << 5;   // 32 rows per wave
  const int col_start = cs << 11;              // 2048 columns per strip
  const int nl = lane & 15;                    // A: row-in-tile, B/C: col
  const int hi = lane >> 4;                    // K-half / M-half selector

  // ---- A fragments (ISA 7.12.2 16x32 bf16), two 16-row stripes.
  Frag A0[8], A1[8];
  {
    const unsigned short* pa0 = feats + (size_t)(row_base + nl) * 256 + hi * 8;
    const unsigned short* pa1 = pa0 + 16 * 256;
    #pragma unroll
    for (int s = 0; s < 8; ++s) {
      A0[s].u[0] = *(const u32x4*)(pa0 + s * 32);
      A0[s].u[1] = *(const u32x4*)(pa0 + s * 32 + 16);
      A1[s].u[0] = *(const u32x4*)(pa1 + s * 32);
      A1[s].u[1] = *(const u32x4*)(pa1 + s * 32 + 16);
    }
  }

  float sums[16];
  float posacc[16];
  #pragma unroll
  for (int q = 0; q < 16; ++q) { sums[q] = 0.f; posacc[q] = -3.0e38f; }

  const float CEXP = 20.609929155556664f;      // (1/0.07)*log2(e)
  const unsigned long long gbase = (unsigned long long)(size_t)feats;
  const unsigned long long gcol0 = gbase + (unsigned long long)col_start * 512ull;

#if HAVE_TDM
  if (wib == 0) {
    tdm_load_stage((unsigned)(size_t)(void*)&bbuf[0][0], gcol0);
    __builtin_amdgcn_s_wait_tensorcnt(0);
  }
  __syncthreads();
#endif

  for (int st = 0; st < STAGES_PER_BLOCK; ++st) {
    const int cur = st & 1;

#if HAVE_TDM
    if (wib == 0 && st + 1 < STAGES_PER_BLOCK)
      tdm_load_stage((unsigned)(size_t)(void*)&bbuf[cur ^ 1][0],
                     gcol0 + (unsigned long long)(st + 1) * (STAGE_ROWS * 512ull));
#else
    {
      const u32x4* src =
          (const u32x4*)(feats + ((size_t)col_start + (size_t)st * STAGE_ROWS) * 256);
      #pragma unroll
      for (int i = threadIdx.x; i < STAGE_ROWS * 32; i += 256) {
        int r = i >> 5, ch = i & 31;
        *(u32x4*)&bbuf[cur][r * LDS_ROW + ch * 8] = src[i];
      }
      __syncthreads();
    }
#endif

    #pragma unroll
    for (int sub = 0; sub < 4; ++sub) {        // 4 x 16-column sub-tiles
      // B fragment: lane = col nl, K offset hi*8 (+16); 528B row stride in LDS ->
      // lane nl hits banks 4nl..4nl+3: conflict-free ds_load_b128 pairs.
      const unsigned short* pb =
          &bbuf[cur][(sub * 16 + nl) * LDS_ROW + hi * 8];

      Frag Bc, Bn;
      Bc.u[0] = *(const u32x4*)(pb);
      Bc.u[1] = *(const u32x4*)(pb + 16);
      v8f acc0 = {0.f, 0.f, 0.f, 0.f, 0.f, 0.f, 0.f, 0.f};
      v8f acc1 = {0.f, 0.f, 0.f, 0.f, 0.f, 0.f, 0.f, 0.f};
      #pragma unroll
      for (int s = 0; s < 8; ++s) {
        if (s < 7) {                           // rotate: prefetch next k-step
          Bn.u[0] = *(const u32x4*)(pb + (s + 1) * 32);
          Bn.u[1] = *(const u32x4*)(pb + (s + 1) * 32 + 16);
        }
        acc0 = __builtin_amdgcn_wmma_f32_16x16x32_bf16(
            false, A0[s].v, false, Bc.v, (short)0, acc0, false, false);
        acc1 = __builtin_amdgcn_wmma_f32_16x16x32_bf16(
            false, A1[s].v, false, Bc.v, (short)0, acc1, false, false);
        Bc = Bn;
      }

      // epilogue: slot q = set*8+v -> row off = set*16 + v + 8*hi
      const int colb = col_start + (st << 6) + (sub << 4);
      const int c    = colb + nl;
      const int d0   = c - row_base - (hi << 3);   // j - i for set0, v=0
      const int e0   = (c ^ 4096) - row_base - (hi << 3); // pos hit iff == off-8hi.. (v)
      #pragma unroll
      for (int set = 0; set < 2; ++set) {
        #pragma unroll
        for (int v = 0; v < 8; ++v) {
          const int q  = set * 8 + v;
          float a      = set ? acc1[v] : acc0[v];
          int   diff   = d0 - (set * 16 + v);  // j - i
          int   u      = diff ^ (diff >> 31);  // diff>=0 ? diff : -1-diff
          u            = u < 20 ? u : 20;
          float w      = (float)u * 0.05f;     // /20
          float e      = exp2f((a - 1.0f) * CEXP);  // exp(sim - S)
          sums[q]      = fmaf(w, e, sums[q]);
          posacc[q]    = (e0 == set * 16 + v) ? a : posacc[q];
        }
      }
    }

#if HAVE_TDM
    if (wib == 0) __builtin_amdgcn_s_wait_tensorcnt(0);
    __syncthreads();
#else
    __syncthreads();
#endif
  }

  // reduce across the 16 lanes sharing each row (xor masks keep the hi half)
  #pragma unroll
  for (int q = 0; q < 16; ++q) {
    float s = sums[q];
    float p = posacc[q];
    #pragma unroll
    for (int m = 8; m >= 1; m >>= 1) {
      s += __shfl_xor(s, m, 32);
      p  = fmaxf(p, __shfl_xor(p, m, 32));
    }
    sums[q]   = s;
    posacc[q] = p;
  }

  if (nl == 0) {                               // lanes 0 and 16: 16 rows each
    #pragma unroll
    for (int q = 0; q < 16; ++q) {
      const int set = q >> 3, v = q & 7;
      const int r   = row_base + set * 16 + v + (hi << 3);
      parts[cs * 8192 + r] = sums[q];          // unique writer per (row, strip)
      if (posacc[q] > -1.0e30f) rowpos[r] = posacc[q];  // unique writer per row
    }
  }
}

// ---------------------------------------------------------------------------
// Kernel 3: final per-row loss + mean. Single block -> deterministic output.
// ---------------------------------------------------------------------------
__global__ __launch_bounds__(256) void supcon_finish(
    const float* __restrict__ rowpos,          // [8192]
    const float* __restrict__ parts,           // [CSPLIT][8192]
    float* __restrict__ out)
{
  __shared__ float red[256];
  const int t = threadIdx.x;
  const float S = 14.285714285714286f;         // 1/0.07
  float acc = 0.f;
  for (int i = t; i < 8192; i += 256) {
    float s = parts[i] + parts[8192 + i] + parts[2 * 8192 + i] + parts[3 * 8192 + i];
    float a = rowpos[i];
    acc += S - a * S + __logf(s);              // loss_i (TEMP/BASE_TEMP = 1)
  }
  red[t] = acc;
  __syncthreads();
  #pragma unroll
  for (int step = 128; step >= 1; step >>= 1) {
    if (t < step) red[t] += red[t + step];
    __syncthreads();
  }
  if (t == 0) out[0] = red[0] * (1.0f / 8192.0f);
}

// ---------------------------------------------------------------------------
extern "C" void kernel_launch(void* const* d_in, const int* in_sizes, int n_in,
                              void* d_out, int out_size, void* d_ws, size_t ws_size,
                              hipStream_t stream) {
  const float* emb = (const float*)d_in[0];        // [16,256,512] fp32
  unsigned short* feats = (unsigned short*)d_ws;   // 4 MB
  float* rowpos = (float*)((char*)d_ws + 8192 * 256 * 2);   // 32 KB
  float* parts  = rowpos + 8192;                            // 128 KB
  float* out = (float*)d_out;                      // scalar fp32

  supcon_normalize<<<256, 256, 0, stream>>>(emb, feats);
  supcon_wmma<<<128, 256, 0, stream>>>(feats, rowpos, parts);
  supcon_finish<<<1, 256, 0, stream>>>(rowpos, parts, out);
}
